// FrameSimilarity_2748779070177
// MI455X (gfx1250) — matrix operand order repaired and checked by reading, not compiled, not run on hardware
//
#include <hip/hip_runtime.h>

typedef _Float16 v8h  __attribute__((ext_vector_type(8)));
typedef _Float16 v16h __attribute__((ext_vector_type(16)));
typedef float    v8f  __attribute__((ext_vector_type(8)));
typedef float    v4f  __attribute__((ext_vector_type(4)));

#define B_   4
#define C_   512
#define T_   1024
#define D_   128
#define L_   101
#define O_   128
#define BT_  (B_ * T_)   // 4096 flattened (b,t) rows

// ---------------------------------------------------------------------------
// WMMA fragment helpers (CDNA5 ISA 7.12.2 layouts, wave32)
// A (16x32 f16): lane L holds row M=L&15; khalf=L>>4; element e -> K =
//   (e<8) ? khalf*8+e : 16+khalf*8+(e-8).  Two contiguous 16B chunks per lane.
// B (32x16 f16): lane L holds column N=L&15; element e -> K = same as A.
//   Two contiguous 16B chunks per lane.
// C/D (16x16 f32): lane L holds col N=L&15; reg r -> row M = r + (L>>4)*8.
// ---------------------------------------------------------------------------
__device__ inline v16h ld_a_frag(const _Float16* __restrict__ row, int khalf) {
  v8h lo = *(const v8h*)(row + khalf * 8);
  v8h hi = *(const v8h*)(row + 16 + khalf * 8);
  v16h a;
#pragma unroll
  for (int i = 0; i < 8; ++i) { a[i] = lo[i]; a[i + 8] = hi[i]; }
  return a;
}

__device__ inline v16h ld_b_frag(const _Float16* __restrict__ row, int khi) {
  v8h lo = *(const v8h*)(row + khi * 8);
  v8h hi = *(const v8h*)(row + 16 + khi * 8);
  v16h b;
#pragma unroll
  for (int i = 0; i < 8; ++i) { b[i] = lo[i]; b[i + 8] = hi[i]; }
  return b;
}

__device__ inline v8f wmma_f16(v16h a, v16h b, v8f c) {
  // emits v_wmma_f32_16x16x32_f16
  return __builtin_amdgcn_wmma_f32_16x16x32_f16(
      /*neg_a=*/false, a, /*neg_b=*/false, b,
      /*c_mod=*/(short)0, c, /*reuse_a=*/false, /*reuse_b=*/false);
}

// ---------------------------------------------------------------------------
// Stage 1: spatial mean over 8x8 = 64 contiguous floats per (b,c,t) tile.
// 8 lanes per tile -> two float4 loads each -> fully coalesced 512 MB stream.
// This is THE bandwidth-bound kernel (~22 us at 23.3 TB/s); everything else
// touches <10 MB.
// x: [B][C][T][64] fp32  ->  feat16: [B][T][C] f16
// ---------------------------------------------------------------------------
__global__ __launch_bounds__(256) void k_reduce_mean(
    const float* __restrict__ x, _Float16* __restrict__ feat16) {
  int tid = threadIdx.x;
  long tile = (long)blockIdx.x * 32 + (tid >> 3);   // (b*C + c)*T + t
  int j = tid & 7;
  const float* base = x + tile * 64;
  // pull the next block's chunk toward the caches (global_prefetch_b8)
  __builtin_prefetch((const char*)base + 32 * 64 * 4, 0, 1);
  const float4* src = (const float4*)base + j * 2;
  float4 a = src[0];
  float4 b = src[1];
  float s = a.x + a.y + a.z + a.w + b.x + b.y + b.z + b.w;
  s += __shfl_xor(s, 1, 32);
  s += __shfl_xor(s, 2, 32);
  s += __shfl_xor(s, 4, 32);
  if (j == 0) {
    long t  = tile % T_;
    long bc = tile / T_;
    long c  = bc % C_;
    long bb = bc / C_;
    feat16[(bb * T_ + t) * C_ + c] = (_Float16)(s * (1.0f / 64.0f));
  }
}

// ---------------------------------------------------------------------------
// Stage 2 prep: convert weights to f16. W_fc is padded from 101 to 128 cols
// with zeros so the fc GEMM can run K=128 in four K32 WMMA steps.
// ---------------------------------------------------------------------------
__global__ __launch_bounds__(256) void k_prep(
    const float* __restrict__ Wproj, const float* __restrict__ Wfc,
    _Float16* __restrict__ Wp16, _Float16* __restrict__ Wfc16) {
  int i = blockIdx.x * 256 + threadIdx.x;
  if (i < D_ * C_) {
    Wp16[i] = (_Float16)Wproj[i];
  } else {
    int k = i - D_ * C_;
    if (k < O_ * 128) {
      int o = k >> 7, l = k & 127;
      float v = (l < L_) ? Wfc[o * L_ + l] : 0.0f;
      Wfc16[o * 128 + l] = (_Float16)v;
    }
  }
}

// ---------------------------------------------------------------------------
// Stage 3: projection p = feat(4096x512) x W_proj^T(512x128), fp32 out.
// One wave per 16x16 output tile, 16 K32 WMMA steps.
// ---------------------------------------------------------------------------
__global__ __launch_bounds__(128) void k_proj(
    const _Float16* __restrict__ feat16, const _Float16* __restrict__ Wp16,
    float* __restrict__ p32) {
  int lane  = threadIdx.x & 31;
  int w     = blockIdx.x * 4 + (threadIdx.x >> 5);
  int mtile = w >> 3;                 // 0..255 (16 rows of flattened b*t)
  int ntile = w & 7;                  // 0..7   (16 of 128 d-columns)
  int ml = lane & 15, khalf = lane >> 4;
  const _Float16* arow = feat16 + (long)(mtile * 16 + ml) * C_;
  const _Float16* brow = Wp16 + (long)(ntile * 16 + ml) * C_;
  v8f acc = {};
#pragma unroll 4
  for (int kk = 0; kk < 16; ++kk) {
    v16h a = ld_a_frag(arow + kk * 32, khalf);
    v16h b = ld_b_frag(brow + kk * 32, khalf);
    acc = wmma_f16(a, b, acc);
  }
#pragma unroll
  for (int r = 0; r < 8; ++r) {
    int row = mtile * 16 + r + khalf * 8;
    p32[(long)row * D_ + ntile * 16 + ml] = acc[r];
  }
}

// ---------------------------------------------------------------------------
// Stage 4: row L2-normalize p -> pn16 (f16). One wave per row.
// ---------------------------------------------------------------------------
__global__ __launch_bounds__(256) void k_norm(
    const float* __restrict__ p32, _Float16* __restrict__ pn16) {
  int lane = threadIdx.x & 31;
  int row  = blockIdx.x * 8 + (threadIdx.x >> 5);
  const float* pr = p32 + (long)row * D_;
  v4f v = *(const v4f*)(pr + lane * 4);
  float ss = v[0] * v[0] + v[1] * v[1] + v[2] * v[2] + v[3] * v[3];
#pragma unroll
  for (int m = 16; m >= 1; m >>= 1) ss += __shfl_xor(ss, m, 32);
  float scale = 1.0f / fmaxf(sqrtf(ss), 1e-12f);
  _Float16* o = pn16 + (long)row * D_ + lane * 4;
  o[0] = (_Float16)(v[0] * scale);
  o[1] = (_Float16)(v[1] * scale);
  o[2] = (_Float16)(v[2] * scale);
  o[3] = (_Float16)(v[3] * scale);
}

// ---------------------------------------------------------------------------
// Stage 5: banded self-similarity + fc + bias + relu.
// One wave per 16-row t-tile. Compute a 16x128 sim strip for s in
// [t0-56, t0+72) (covers the +/-50 band; out-of-batch s -> zero fragments,
// matching the reference's zero padding), spill to LDS (in-order per wave),
// gather band[r][l] = strip[r][r+l+6] (l<101, else 0) as f16 A fragments,
// then band(16x128) x W_fc16^T(128x128) via WMMA, + b_fc, relu.
// ---------------------------------------------------------------------------
__global__ __launch_bounds__(128) void k_band_fc(
    const _Float16* __restrict__ pn16, const _Float16* __restrict__ Wfc16,
    const float* __restrict__ bfc, float* __restrict__ out) {
  __shared__ float strip[4][16 * 128];
  int lane = threadIdx.x & 31;
  int wv   = threadIdx.x >> 5;
  int tile = blockIdx.x * 4 + wv;         // 0..255
  int gt0  = tile * 16;                   // flattened (b,t) row base
  int b    = gt0 >> 10;
  int tl0  = gt0 & 1023;
  int ml = lane & 15, khalf = lane >> 4;
  float* st = strip[wv];

  // A fragments: pn rows gt0..gt0+15, full K=128 (4 frags, reused 8x)
  const _Float16* arow = pn16 + (long)(gt0 + ml) * D_;
  v16h afr[4];
#pragma unroll
  for (int kk = 0; kk < 4; ++kk) afr[kk] = ld_a_frag(arow + kk * 32, khalf);

  int sb = tl0 - 56;                       // strip column 0 -> local time sb
  v16h zero = {};
#pragma unroll
  for (int nt = 0; nt < 8; ++nt) {
    v8f acc = {};
    int s = sb + nt * 16 + ml;             // this lane's B column (pn row)
    bool ok = (s >= 0) && (s < T_);
    const _Float16* brow = pn16 + ((long)b * T_ + (ok ? s : 0)) * D_;
#pragma unroll
    for (int kk = 0; kk < 4; ++kk) {
      v16h bf = ld_b_frag(brow + kk * 32, khalf);
      if (!ok) bf = zero;
      acc = wmma_f16(afr[kk], bf, acc);
    }
#pragma unroll
    for (int r = 0; r < 8; ++r)
      st[(r + khalf * 8) * 128 + nt * 16 + ml] = acc[r];
  }
  __syncthreads();   // LDS ops are in-order per wave; barrier for safety

  // band A fragments from LDS: band[r][l] = strip[r][r + l + 6]
  v16h bnd[4];
#pragma unroll
  for (int kk = 0; kk < 4; ++kk) {
    v16h f;
#pragma unroll
    for (int e = 0; e < 16; ++e) {
      int l = kk * 32 + (e < 8 ? khalf * 8 + e : 16 + khalf * 8 + (e - 8));
      float v = (l < L_) ? st[ml * 128 + ml + l + 6] : 0.0f;
      f[e] = (_Float16)v;
    }
    bnd[kk] = f;
  }

  // fc: out tile 16x128 = band(16x128) x Wfc16^T, + bias, relu
#pragma unroll
  for (int ot = 0; ot < 8; ++ot) {
    v8f acc = {};
    int o = ot * 16 + ml;
    const _Float16* brow = Wfc16 + (long)o * 128;
#pragma unroll
    for (int kk = 0; kk < 4; ++kk)
      acc = wmma_f16(bnd[kk], ld_b_frag(brow + kk * 32, khalf), acc);
    float bias = bfc[o];
#pragma unroll
    for (int r = 0; r < 8; ++r) {
      int row = gt0 + r + khalf * 8;
      out[(long)row * O_ + o] = fmaxf(acc[r] + bias, 0.0f);
    }
  }
}

// ---------------------------------------------------------------------------
extern "C" void kernel_launch(void* const* d_in, const int* in_sizes, int n_in,
                              void* d_out, int out_size, void* d_ws, size_t ws_size,
                              hipStream_t stream) {
  const float* x     = (const float*)d_in[0];   // (4,512,1024,8,8)
  const float* Wproj = (const float*)d_in[1];   // (128,512)
  const float* Wfc   = (const float*)d_in[2];   // (128,101)
  const float* bfc   = (const float*)d_in[3];   // (128,)
  float* out = (float*)d_out;                   // (4,1024,128)

  char* ws = (char*)d_ws;                       // ~7.5 MB total scratch
  _Float16* feat16 = (_Float16*)ws; ws += (size_t)BT_ * C_ * 2;   // 4 MB
  _Float16* Wp16   = (_Float16*)ws; ws += (size_t)D_ * C_ * 2;    // 128 KB
  _Float16* Wfc16  = (_Float16*)ws; ws += (size_t)O_ * 128 * 2;   // 32 KB
  float*    p32    = (float*)ws;    ws += (size_t)BT_ * D_ * 4;   // 2 MB
  _Float16* pn16   = (_Float16*)ws; ws += (size_t)BT_ * D_ * 2;   // 1 MB

  k_prep<<<(D_ * C_ + O_ * 128 + 255) / 256, 256, 0, stream>>>(Wproj, Wfc, Wp16, Wfc16);
  k_reduce_mean<<<(B_ * C_ * T_) / 32, 256, 0, stream>>>(x, feat16);
  k_proj<<<(BT_ / 16) * 8 / 4, 128, 0, stream>>>(feat16, Wp16, p32);   // 512 blocks
  k_norm<<<BT_ / 8, 256, 0, stream>>>(p32, pn16);                       // 512 blocks
  k_band_fc<<<(BT_ / 16) / 4, 128, 0, stream>>>(pn16, Wfc16, bfc, out); // 64 blocks
}
